// DiagTrainableLDAHead_54116587929770
// MI455X (gfx1250) — matrix operand order repaired and checked by reading, not compiled, not run on hardware
//
#include <hip/hip_runtime.h>
#include <hip/hip_bf16.h>
#include <stdint.h>

// ---------------------------------------------------------------------------
// DiagTrainableLDAHead on MI455X (gfx1250, wave32, WMMA)
// out[n,c] = log_prior[c] - 0.5*(z_sq[n] - 2*cross[n,c] + mu_sq[c] + log_det)
// cross = (z*inv_var) @ mu^T  via bf16x3 split-precision WMMA (16x16x32 bf16),
// staged through LDS with GLOBAL_LOAD_ASYNC_TO_LDS_B128 (ASYNCcnt) double
// buffering -> no VGPR staging, no scratch spills.
// ---------------------------------------------------------------------------

#define N_ROWS 8192
#define C_CLS  2048
#define D_DIM  512

#define BM 128
#define BN 64
#define BK 32
#define LDSROW 40          // bf16 elems per LDS row (32 data + 8 pad) -> conflict-free

typedef __attribute__((ext_vector_type(16))) __bf16 v16bf;
typedef __attribute__((ext_vector_type(8)))  float  v8f;

// ---------------- workspace layout (bytes) ----------------
#define WS_INV_VAR   0u                       // 512 f32
#define WS_LOG_PRIOR 2048u                    // 2048 f32
#define WS_Z_SQ      10240u                   // 8192 f32
#define WS_MU_SQ     43008u                   // 2048 f32
#define WS_LOG_DET   51200u                   // 1 f32
#define WS_AHI       65536u                   // N*D bf16 (8 MB)
#define WS_ALO       (WS_AHI + 8388608u)      // 8 MB
#define WS_BHI       (WS_ALO + 8388608u)      // C*D bf16 (2 MB)
#define WS_BLO       (WS_BHI + 2097152u)      // 2 MB  (total ~21 MB)

// ---------------------------------------------------------------------------
// CDNA5 async copy helpers
// ---------------------------------------------------------------------------
__device__ __forceinline__ void async_b128(uint16_t* lds_dst, const uint16_t* g_src) {
    // LDS operand = low 32 bits of the generic shared-space address (= LDS byte offset).
    asm volatile("global_load_async_to_lds_b128 %0, %1, off"
                 :: "v"((uint32_t)(uintptr_t)lds_dst), "v"(g_src)
                 : "memory");
}
__device__ __forceinline__ void wait_async0() {
#if __has_builtin(__builtin_amdgcn_s_wait_asynccnt)
    __builtin_amdgcn_s_wait_asynccnt(0);
#else
    asm volatile("s_wait_asynccnt 0x0" ::: "memory");
#endif
}

// ---------------------------------------------------------------------------
// Pre-pass 1: inv_var, log_det, log_softmax(prior_logits). One block.
// ---------------------------------------------------------------------------
__global__ __launch_bounds__(256)
void lda_prep_scalars(const float* __restrict__ log_cov_diag,
                      const float* __restrict__ prior_logits,
                      float* __restrict__ inv_var,
                      float* __restrict__ log_prior,
                      float* __restrict__ log_det_out) {
    __shared__ float red[256];
    const int t = threadIdx.x;

    float ld = 0.f;
    for (int d = t; d < D_DIM; d += 256) {
        float lcd = log_cov_diag[d];
        inv_var[d] = __expf(-lcd);
        ld += lcd;
    }
    red[t] = ld; __syncthreads();
    for (int s = 128; s > 0; s >>= 1) { if (t < s) red[t] += red[t + s]; __syncthreads(); }
    if (t == 0) *log_det_out = red[0];
    __syncthreads();

    float m = -INFINITY;
    for (int c = t; c < C_CLS; c += 256) m = fmaxf(m, prior_logits[c]);
    red[t] = m; __syncthreads();
    for (int s = 128; s > 0; s >>= 1) { if (t < s) red[t] = fmaxf(red[t], red[t + s]); __syncthreads(); }
    m = red[0]; __syncthreads();

    float acc = 0.f;
    for (int c = t; c < C_CLS; c += 256) acc += __expf(prior_logits[c] - m);
    red[t] = acc; __syncthreads();
    for (int s = 128; s > 0; s >>= 1) { if (t < s) red[t] += red[t + s]; __syncthreads(); }
    float lse = m + __logf(red[0]);
    __syncthreads();

    for (int c = t; c < C_CLS; c += 256) log_prior[c] = prior_logits[c] - lse;
}

// ---------------------------------------------------------------------------
// Pre-pass 2: split rows into bf16 hi/lo, compute sum(x^2 * inv_var) per row.
// ---------------------------------------------------------------------------
__global__ __launch_bounds__(256)
void lda_split(const float* __restrict__ X,
               const float* __restrict__ inv_var,
               uint16_t* __restrict__ Hi, uint16_t* __restrict__ Lo,
               float* __restrict__ sq, int premul) {
    const int lane = threadIdx.x & 31;
    const int wave = threadIdx.x >> 5;
    const int row  = blockIdx.x * 8 + wave;
    const float* xr = X + (size_t)row * D_DIM;

    float acc = 0.f;
#pragma unroll
    for (int j = 0; j < D_DIM / 32; ++j) {
        const int d = lane + j * 32;
        const float x  = xr[d];
        const float iv = inv_var[d];
        acc += x * x * iv;
        const float a = premul ? x * iv : x;
        const uint32_t ab = __float_as_uint(a);
        const uint16_t hb = (uint16_t)(ab >> 16);                 // truncate to bf16
        const float    hf = __uint_as_float((uint32_t)hb << 16);
        const float    lo = a - hf;                                // exact residual
        const uint16_t lb = (uint16_t)(__float_as_uint(lo) >> 16);
        Hi[(size_t)row * D_DIM + d] = hb;
        Lo[(size_t)row * D_DIM + d] = lb;
    }
#pragma unroll
    for (int m = 16; m >= 1; m >>= 1) acc += __shfl_xor(acc, m, 32);
    if (lane == 0) sq[row] = acc;
}

// ---------------------------------------------------------------------------
// Main bf16x3 WMMA GEMM + fused epilogue.
// Block = 256 threads (8 waves, 4(M) x 2(N)); tile 128x64; K-step 32.
// LDS double buffered, filled by async global->LDS DMA.
// ---------------------------------------------------------------------------
__device__ __forceinline__ v16bf frag_a(const uint16_t* s, int row, int kh) {
    // A 16x32 bf16: lane<16 -> K {0..7, 16..23}; lane>=16 -> K {8..15, 24..31}
    union { uint4 q[2]; v16bf v; } u;
    const uint16_t* p = s + row * LDSROW;
    u.q[0] = *(const uint4*)(p + kh * 8);
    u.q[1] = *(const uint4*)(p + 16 + kh * 8);
    return u.v;
}
__device__ __forceinline__ v16bf frag_b(const uint16_t* s, int row, int kh) {
    // B 32x16 bf16: lane<16 -> K {0..7, 16..23}; lane>=16 -> K {8..15, 24..31}
    union { uint4 q[2]; v16bf v; } u;
    const uint16_t* p = s + row * LDSROW;
    u.q[0] = *(const uint4*)(p + kh * 8);
    u.q[1] = *(const uint4*)(p + 16 + kh * 8);
    return u.v;
}

__global__ __launch_bounds__(256)
void lda_wmma_gemm(const uint16_t* __restrict__ Ahi, const uint16_t* __restrict__ Alo,
                   const uint16_t* __restrict__ Bhi, const uint16_t* __restrict__ Blo,
                   const float* __restrict__ z_sq,  const float* __restrict__ mu_sq,
                   const float* __restrict__ log_prior, const float* __restrict__ log_det_p,
                   float* __restrict__ out) {
    __shared__ uint16_t sAh[2][BM * LDSROW], sAl[2][BM * LDSROW];   // 2*2*10240 B
    __shared__ uint16_t sBh[2][BN * LDSROW], sBl[2][BN * LDSROW];   // 2*2*5120 B  (total 60 KB)

    const int t    = threadIdx.x;
    const int lane = t & 31;
    const int wave = t >> 5;
    const int wm   = wave & 3;          // 4 waves along M (32 rows each)
    const int wn   = wave >> 2;         // 2 waves along N (32 cols each)
    const int n0   = blockIdx.y * BM;
    const int c0   = blockIdx.x * BN;
    const int kh   = lane >> 4;
    const int l16  = lane & 15;

    // Per-thread async-copy geometry (constant across K-steps):
    // A slab: 128 rows x 4 chunks(16B) x {hi,lo} -> 2 chunks/thread/array
    // B slab:  64 rows x 4 chunks(16B) x {hi,lo} -> 1 chunk/thread/array
    const int ra0 = t >> 2,        qa0 = t & 3;          // A chunk 0
    const int ra1 = (t + 256) >> 2, qa1 = t & 3;         // A chunk 1
    const int rb  = t >> 2,        qb  = t & 3;          // B chunk (rb in 0..63)

    const size_t gA0 = (size_t)(n0 + ra0) * D_DIM + qa0 * 8;
    const size_t gA1 = (size_t)(n0 + ra1) * D_DIM + qa1 * 8;
    const size_t gB  = (size_t)(c0 + rb ) * D_DIM + qb  * 8;
    const int    lA0 = ra0 * LDSROW + qa0 * 8;
    const int    lA1 = ra1 * LDSROW + qa1 * 8;
    const int    lB  = rb  * LDSROW + qb  * 8;

    v8f acc[2][2];
    const v8f vzero = {0.f, 0.f, 0.f, 0.f, 0.f, 0.f, 0.f, 0.f};
#pragma unroll
    for (int mt = 0; mt < 2; ++mt)
#pragma unroll
        for (int nt = 0; nt < 2; ++nt) acc[mt][nt] = vzero;

    constexpr int S = D_DIM / BK;       // 16 K-steps

    // Issue async DMA for the first K-slab into buffer 0.
    {
        const int k0 = 0;
        async_b128(&sAh[0][lA0], Ahi + gA0 + k0);
        async_b128(&sAl[0][lA0], Alo + gA0 + k0);
        async_b128(&sAh[0][lA1], Ahi + gA1 + k0);
        async_b128(&sAl[0][lA1], Alo + gA1 + k0);
        async_b128(&sBh[0][lB],  Bhi + gB  + k0);
        async_b128(&sBl[0][lB],  Blo + gB  + k0);
    }

    for (int k = 0; k < S; ++k) {
        const int cur = k & 1;
        wait_async0();                  // own DMA batch for buf[cur] complete
        __syncthreads();                // -> every wave's batch complete, data visible

        if (k + 1 < S) {                // overlap next slab's DMA with the WMMA work
            const int nxt = cur ^ 1;
            const int k0  = (k + 1) * BK;
            async_b128(&sAh[nxt][lA0], Ahi + gA0 + k0);
            async_b128(&sAl[nxt][lA0], Alo + gA0 + k0);
            async_b128(&sAh[nxt][lA1], Ahi + gA1 + k0);
            async_b128(&sAl[nxt][lA1], Alo + gA1 + k0);
            async_b128(&sBh[nxt][lB],  Bhi + gB  + k0);
            async_b128(&sBl[nxt][lB],  Blo + gB  + k0);
        }

        v16bf ah[2], al[2];
#pragma unroll
        for (int mt = 0; mt < 2; ++mt) {
            const int r = wm * 32 + mt * 16 + l16;
            ah[mt] = frag_a(sAh[cur], r, kh);
            al[mt] = frag_a(sAl[cur], r, kh);
        }
#pragma unroll
        for (int nt = 0; nt < 2; ++nt) {
            const int rc = wn * 32 + nt * 16 + l16;
            const v16bf bh = frag_b(sBh[cur], rc, kh);
            const v16bf bl = frag_b(sBl[cur], rc, kh);
#pragma unroll
            for (int mt = 0; mt < 2; ++mt) {
                acc[mt][nt] = __builtin_amdgcn_wmma_f32_16x16x32_bf16(
                    false, ah[mt], false, bh, (short)0, acc[mt][nt], false, false);
                acc[mt][nt] = __builtin_amdgcn_wmma_f32_16x16x32_bf16(
                    false, ah[mt], false, bl, (short)0, acc[mt][nt], false, false);
                acc[mt][nt] = __builtin_amdgcn_wmma_f32_16x16x32_bf16(
                    false, al[mt], false, bh, (short)0, acc[mt][nt], false, false);
            }
        }
        __syncthreads();                // buf[cur] free before iter k+1 DMAs into it
    }

    // Fused epilogue: out = cross + log_prior[c] - 0.5*(z_sq[n] + mu_sq[c] + log_det)
    const float log_det = *log_det_p;
#pragma unroll
    for (int nt = 0; nt < 2; ++nt) {
        const int c = c0 + wn * 32 + nt * 16 + l16;
        const float sc = log_prior[c] - 0.5f * (mu_sq[c] + log_det);
#pragma unroll
        for (int mt = 0; mt < 2; ++mt) {
            const int nbase = n0 + wm * 32 + mt * 16 + kh * 8;
#pragma unroll
            for (int v = 0; v < 8; ++v) {
                const int n = nbase + v;
                out[(size_t)n * C_CLS + c] = acc[mt][nt][v] + sc - 0.5f * z_sq[n];
            }
        }
    }
}

// ---------------------------------------------------------------------------
extern "C" void kernel_launch(void* const* d_in, const int* in_sizes, int n_in,
                              void* d_out, int out_size, void* d_ws, size_t ws_size,
                              hipStream_t stream) {
    const float* z   = (const float*)d_in[0];  // [N, D]
    const float* mu  = (const float*)d_in[1];  // [C, D]
    const float* lcd = (const float*)d_in[2];  // [D]
    const float* pl  = (const float*)d_in[3];  // [C]
    float* out = (float*)d_out;                // [N, C]

    char* ws = (char*)d_ws;
    float*    inv_var   = (float*)(ws + WS_INV_VAR);
    float*    log_prior = (float*)(ws + WS_LOG_PRIOR);
    float*    z_sq      = (float*)(ws + WS_Z_SQ);
    float*    mu_sq     = (float*)(ws + WS_MU_SQ);
    float*    log_det   = (float*)(ws + WS_LOG_DET);
    uint16_t* A_hi      = (uint16_t*)(ws + WS_AHI);
    uint16_t* A_lo      = (uint16_t*)(ws + WS_ALO);
    uint16_t* B_hi      = (uint16_t*)(ws + WS_BHI);
    uint16_t* B_lo      = (uint16_t*)(ws + WS_BLO);

    lda_prep_scalars<<<1, 256, 0, stream>>>(lcd, pl, inv_var, log_prior, log_det);
    lda_split<<<N_ROWS / 8, 256, 0, stream>>>(z,  inv_var, A_hi, A_lo, z_sq, 1);
    lda_split<<<C_CLS / 8, 256, 0, stream>>>(mu, inv_var, B_hi, B_lo, mu_sq, 0);

    dim3 grid(C_CLS / BN, N_ROWS / BM);   // 32 x 64 tiles
    lda_wmma_gemm<<<grid, 256, 0, stream>>>(A_hi, A_lo, B_hi, B_lo,
                                            z_sq, mu_sq, log_prior, log_det, out);
}
